// EfficientWindowAttention_42228118454705
// MI455X (gfx1250) — matrix operand order repaired and checked by reading, not compiled, not run on hardware
//
#include <hip/hip_runtime.h>
#include <hip/hip_bf16.h>

typedef _Float16 h16;
typedef __attribute__((ext_vector_type(16))) _Float16 v16h;
typedef __attribute__((ext_vector_type(8)))  float    v8f;

// ---------------- problem constants (fixed by the reference) ----------------
#define BATCH 8
#define CH    64          // channels
#define HW    256         // H == W
#define WS    8           // window size
#define NTOK  64          // tokens per window (WS*WS)
#define NH    4           // heads
#define HD    16          // head dim
#define NWIN  8192        // total windows
#define EPS   1e-5f

// ---------------- LDS layout (bytes); dynamic shared memory ----------------
// f16 matrices: row stride 72 halves (144 B, 16B-aligned, bank-spread).
// f32 matrices: row stride 68 floats (272 B, 16B-aligned, bank-spread).
#define OFF_WQ   0                         // 192 x 72 h  = 27648 B  (qkv_w  f16, [o][c])
#define OFF_WP   27648                     //  64 x 72 h  =  9216 B  (proj_w f16, [o][c])
#define OFF_RPB  36864                     // 225 x 4 f32 =  3600 B
#define OFF_NW   40464                     //  64 f32
#define OFF_NB   40720                     //  64 f32
#define OFF_X    40976                     //  64 x 72 h  =  9216 B  (normalized X f16)
#define OFF_QKV  50192                     //  64 x 200 h = 25600 B  (QKV output f16)
#define OFF_S    75792                     // 4 x 64 x 68 f32 = 69632 B (scores / Xraw / Z)
#define OFF_O    145424                    //  64 x 72 h  =  9216 B  (pre-proj attn out f16)
#define SMEM_BYTES 154640

#define WQ_STR  72
#define WP_STR  72
#define X_STR   72
#define QKV_STR 200
#define S_STR   68
#define O_STR   72

// ---------------- WMMA fragment loaders (contiguous k order, same for A and B) ----------------
// A-matrix 16x32 f16: lane = m(0..15) | khalf(bit4). VGPR v holds
// k = 16*khalf + 2v + {0,1}. Source is row-major [m][k].
__device__ __forceinline__ v16h load_frag_a(const h16* base, int stride,
                                            int row0, int col0, int kvalid) {
  const int lane = threadIdx.x & 31;
  const int m  = lane & 15;
  const int kh = lane >> 4;
  const h16* rp = base + (row0 + m) * stride + col0;
  v16h a;
#pragma unroll
  for (int v = 0; v < 8; ++v) {
    const int kb = kh * 16 + 2 * v;
    a[2 * v]     = (kb     < kvalid) ? rp[kb]     : (h16)0.f;
    a[2 * v + 1] = (kb + 1 < kvalid) ? rp[kb + 1] : (h16)0.f;
  }
  return a;
}

// B-matrix 32x16 f16: lane = n(0..15) | khalf(bit4). VGPR v holds
// k = 16*khalf + 2v + {0,1}. Source stored row-major as [n][k] (e.g. W[o][c]).
__device__ __forceinline__ v16h load_frag_b_nk(const h16* base, int stride,
                                               int nrow0, int kcol0, int kvalid) {
  const int lane = threadIdx.x & 31;
  const int n  = lane & 15;
  const int kh = lane >> 4;
  const h16* rp = base + (nrow0 + n) * stride + kcol0 + kh * 16;
  v16h b;
#pragma unroll
  for (int v = 0; v < 8; ++v) {
    const int k = kh * 16 + 2 * v;
    b[2 * v]     = (k     < kvalid) ? rp[2 * v]     : (h16)0.f;
    b[2 * v + 1] = (k + 1 < kvalid) ? rp[2 * v + 1] : (h16)0.f;
  }
  return b;
}

// B-matrix where source is row-major [k][n] (e.g. V[token][d]).
__device__ __forceinline__ v16h load_frag_b_kn(const h16* base, int stride,
                                               int krow0, int ncol0, int kvalid) {
  const int lane = threadIdx.x & 31;
  const int n  = lane & 15;
  const int kh = lane >> 4;
  v16h b;
#pragma unroll
  for (int v = 0; v < 8; ++v) {
    const int k = kh * 16 + 2 * v;
    b[2 * v]     = (k     < kvalid) ? base[(krow0 + k)     * stride + ncol0 + n] : (h16)0.f;
    b[2 * v + 1] = (k + 1 < kvalid) ? base[(krow0 + k + 1) * stride + ncol0 + n] : (h16)0.f;
  }
  return b;
}

#define WMMA_F16(A, B, C) \
  __builtin_amdgcn_wmma_f32_16x16x32_f16(false, (A), false, (B), (short)0, (C), false, false)

// ----------------------------------------------------------------------------
__global__ __launch_bounds__(128) void swin_window_attn_kernel(
    const float* __restrict__ x, const float* __restrict__ norm_w,
    const float* __restrict__ norm_b, const float* __restrict__ qkv_w,
    const float* __restrict__ proj_w, const float* __restrict__ attn_scale,
    const float* __restrict__ rpb_table, float* __restrict__ out) {
  extern __shared__ char smem[];
  h16*   sWq   = (h16*)(smem + OFF_WQ);
  h16*   sWp   = (h16*)(smem + OFF_WP);
  float* sRPB  = (float*)(smem + OFF_RPB);
  float* sNW   = (float*)(smem + OFF_NW);
  float* sNB   = (float*)(smem + OFF_NB);
  h16*   sX    = (h16*)(smem + OFF_X);
  h16*   sQKV  = (h16*)(smem + OFF_QKV);
  float* sS    = (float*)(smem + OFF_S);   // 4 head blocks of 64x68 f32
  h16*   sO    = (h16*)(smem + OFF_O);
  float* sXraw = sS;                        // alias: raw window (64x68 f32)
  float* sZ    = sS;                        // alias: proj output (64x68 f32)

  const int tid  = threadIdx.x;
  const int lane = tid & 31;
  const int wave = tid >> 5;
  const float ascale = attn_scale[0];
  const float qk_scl = 0.25f;               // hd^-0.5, hd = 16

  // ---- one-time staging of weights / bias table / LN params ----
  for (int i = tid; i < 192 * CH; i += 128)
    sWq[(i >> 6) * WQ_STR + (i & 63)] = (h16)qkv_w[i];
  for (int i = tid; i < CH * CH; i += 128)
    sWp[(i >> 6) * WP_STR + (i & 63)] = (h16)proj_w[i];
  for (int i = tid; i < 225 * NH; i += 128) sRPB[i] = rpb_table[i];
  if (tid < CH) { sNW[tid] = norm_w[tid]; sNB[tid] = norm_b[tid]; }
  __syncthreads();

  for (int w = blockIdx.x; w < NWIN; w += gridDim.x) {
    const int bb = w >> 10;
    const int wh = (w >> 5) & 31;
    const int ww = w & 31;

    // prefetch next window's tile (global_prefetch_b8 path)
    {
      const int wn = w + gridDim.x;
      if (wn < NWIN) {
        const int bn = wn >> 10, whn = (wn >> 5) & 31, wwn = wn & 31;
        const int c = tid >> 3, r = tid & 7;
        __builtin_prefetch(
            x + ((size_t)(bn * CH + c) * (HW * HW)) +
                (size_t)(whn * WS + r) * HW + wwn * WS, 0, 3);
      }
    }

    // ---- phase 1: async-copy window [C][8x8] -> sXraw[token][channel] ----
    // GVS mode: uniform SGPR base (window origin), per-lane 32-bit byte offset.
    // Each lane moves one dword directly global->LDS (transpose via addressing);
    // tracked on ASYNCcnt, no VGPR round-trip, no ds_store traffic.
    {
      const float* xwin = x + ((size_t)bb * CH) * (HW * HW) +
                          (size_t)(wh * WS) * HW + ww * WS;
      const unsigned long long gbase = (unsigned long long)(uintptr_t)xwin;
      const unsigned ldsbase = (unsigned)(uintptr_t)(void*)sXraw;
#pragma unroll
      for (int i = 0; i < 32; ++i) {
        const int e  = tid + i * 128;          // 0..4095 elements
        const int cc = e & 7, r = (e >> 3) & 7, c = e >> 6;
        const unsigned goff  = (unsigned)((c * (HW * HW) + r * HW + cc) * 4);
        const unsigned laddr =
            ldsbase + (unsigned)(((r * WS + cc) * S_STR + c) * 4);
        asm volatile("global_load_async_to_lds_b32 %0, %1, %2"
                     :: "v"(laddr), "v"(goff), "s"(gbase) : "memory");
      }
      asm volatile("s_wait_asynccnt 0x0" ::: "memory");
    }
    __syncthreads();

    // ---- phase 2: LayerNorm over channels, emit f16 sX ----
    if (tid < NTOK) {
      const float* xr = sXraw + tid * S_STR;   // 16B-aligned row
      float s = 0.f, s2 = 0.f;
#pragma unroll
      for (int c4 = 0; c4 < CH / 4; ++c4) {
        const float4 v = ((const float4*)xr)[c4];
        s  += v.x + v.y + v.z + v.w;
        s2 += v.x * v.x + v.y * v.y + v.z * v.z + v.w * v.w;
      }
      const float mu  = s * (1.f / 64.f);
      const float var = s2 * (1.f / 64.f) - mu * mu;
      const float inv = rsqrtf(var + EPS);
#pragma unroll 8
      for (int c = 0; c < CH; ++c)
        sX[tid * X_STR + c] = (h16)((xr[c] - mu) * inv * sNW[c] + sNB[c]);
    }
    __syncthreads();

    // ---- phase 3: QKV GEMM  Y[64 x 192] = Xn @ Wqkv^T  (f16 WMMA) ----
#pragma unroll
    for (int nti = 0; nti < 3; ++nti) {
      const int nt = wave * 3 + nti;          // 12 output tiles of 16
#pragma unroll
      for (int mt = 0; mt < 4; ++mt) {
        v8f acc = {};
#pragma unroll
        for (int ks = 0; ks < 2; ++ks) {
          v16h a = load_frag_a(sX, X_STR, mt * 16, ks * 32, 32);
          v16h b = load_frag_b_nk(sWq, WQ_STR, nt * 16, ks * 32, 32);
          acc = WMMA_F16(a, b, acc);
        }
        const int mb = mt * 16 + (lane >> 4) * 8;
        const int n  = nt * 16 + (lane & 15);
#pragma unroll
        for (int v = 0; v < 8; ++v) sQKV[(mb + v) * QKV_STR + n] = (h16)acc[v];
      }
    }
    __syncthreads();

    // ---- phase 4: attention (one head per wave) ----
    float* sSh = sS + wave * (NTOK * S_STR);
    // 4a: S = Q @ K^T   (K-dim = 16, zero-padded to 32 at frag-load time)
#pragma unroll
    for (int mt = 0; mt < 4; ++mt)
#pragma unroll
      for (int nt = 0; nt < 4; ++nt) {
        v16h a = load_frag_a(sQKV, QKV_STR, mt * 16, wave * HD, HD);         // Q
        v16h b = load_frag_b_nk(sQKV, QKV_STR, nt * 16, CH + wave * HD, HD); // K^T
        v8f acc = {};
        acc = WMMA_F16(a, b, acc);
        const int mb = mt * 16 + (lane >> 4) * 8;
        const int n  = nt * 16 + (lane & 15);
#pragma unroll
        for (int v = 0; v < 8; ++v) sSh[(mb + v) * S_STR + n] = acc[v];
      }
    // 4b: softmax with relative position bias (float4-vectorized rows);
    //     repack P as f16 packed at the start of each row.
#pragma unroll
    for (int rr = 0; rr < 2; ++rr) {
      const int q  = lane + rr * 32;
      const int ri = q >> 3, ci = q & 7;
      float* row = sSh + q * S_STR;            // 16B-aligned
      float4* row4 = (float4*)row;
      float mx = -1e30f;
      for (int m4 = 0; m4 < 16; ++m4) {
        const int m  = m4 * 4;
        const int idx0 = (ri - (m >> 3) + 7) * 15 + (ci - (m & 7) + 7);
        const float4 v = row4[m4];
        mx = fmaxf(mx, fmaxf(fmaxf(v.x * qk_scl + sRPB[idx0 * NH + wave],
                                   v.y * qk_scl + sRPB[(idx0 - 1) * NH + wave]),
                             fmaxf(v.z * qk_scl + sRPB[(idx0 - 2) * NH + wave],
                                   v.w * qk_scl + sRPB[(idx0 - 3) * NH + wave])));
      }
      float sum = 0.f;
      for (int m4 = 0; m4 < 16; ++m4) {
        const int m  = m4 * 4;
        const int idx0 = (ri - (m >> 3) + 7) * 15 + (ci - (m & 7) + 7);
        float4 v = row4[m4];
        v.x = __expf(v.x * qk_scl + sRPB[idx0 * NH + wave] - mx);
        v.y = __expf(v.y * qk_scl + sRPB[(idx0 - 1) * NH + wave] - mx);
        v.z = __expf(v.z * qk_scl + sRPB[(idx0 - 2) * NH + wave] - mx);
        v.w = __expf(v.w * qk_scl + sRPB[(idx0 - 3) * NH + wave] - mx);
        row4[m4] = v;
        sum += v.x + v.y + v.z + v.w;
      }
      const float invs = 1.f / sum;
      h16* prow = (h16*)row;                   // f16 P packed at row start
      for (int m4 = 0; m4 < 16; ++m4) {
        const float4 v = row4[m4];             // reads stay ahead of f16 writes
        prow[m4 * 4 + 0] = (h16)(v.x * invs);
        prow[m4 * 4 + 1] = (h16)(v.y * invs);
        prow[m4 * 4 + 2] = (h16)(v.z * invs);
        prow[m4 * 4 + 3] = (h16)(v.w * invs);
      }
    }
    // 4c: O_h = P @ V   ([64x64] @ [64x16])
    const h16* pbase = (const h16*)sSh;        // P rows, stride 2*S_STR halves
#pragma unroll
    for (int mt = 0; mt < 4; ++mt) {
      v8f acc = {};
#pragma unroll
      for (int ks = 0; ks < 2; ++ks) {
        v16h a = load_frag_a(pbase, 2 * S_STR, mt * 16, ks * 32, 32);
        v16h b = load_frag_b_kn(sQKV, QKV_STR, ks * 32, 2 * CH + wave * HD, 32);
        acc = WMMA_F16(a, b, acc);
      }
      const int mb = mt * 16 + (lane >> 4) * 8;
      const int n  = wave * HD + (lane & 15);
#pragma unroll
      for (int v = 0; v < 8; ++v) sO[(mb + v) * O_STR + n] = (h16)acc[v];
    }
    __syncthreads();

    // ---- phase 5: proj  Z[64x64] = O @ Wp^T  (wave owns one 16-row strip) ----
    {
      const int mt = wave;
#pragma unroll
      for (int nt = 0; nt < 4; ++nt) {
        v8f acc = {};
#pragma unroll
        for (int ks = 0; ks < 2; ++ks) {
          v16h a = load_frag_a(sO, O_STR, mt * 16, ks * 32, 32);
          v16h b = load_frag_b_nk(sWp, WP_STR, nt * 16, ks * 32, 32);
          acc = WMMA_F16(a, b, acc);
        }
        const int mb = mt * 16 + (lane >> 4) * 8;
        const int n  = nt * 16 + (lane & 15);
#pragma unroll
        for (int v = 0; v < 8; ++v) sZ[(mb + v) * S_STR + n] = acc[v];
      }
    }
    __syncthreads();

    // ---- phase 6: residual write  out = x + attn_scale * Z ----
#pragma unroll
    for (int i = 0; i < 4; ++i) {
      const int seg = tid + i * 128;
      const int c = seg >> 3, r = seg & 7;
      const size_t off = ((size_t)(bb * CH + c) * (HW * HW)) +
                         (size_t)(wh * WS + r) * HW + ww * WS;
      const float4 a0 = ((const float4*)(x + off))[0];
      const float4 a1 = ((const float4*)(x + off))[1];
      const int t0 = r * WS;
      float4 o0, o1;
      o0.x = a0.x + ascale * sZ[(t0 + 0) * S_STR + c];
      o0.y = a0.y + ascale * sZ[(t0 + 1) * S_STR + c];
      o0.z = a0.z + ascale * sZ[(t0 + 2) * S_STR + c];
      o0.w = a0.w + ascale * sZ[(t0 + 3) * S_STR + c];
      o1.x = a1.x + ascale * sZ[(t0 + 4) * S_STR + c];
      o1.y = a1.y + ascale * sZ[(t0 + 5) * S_STR + c];
      o1.z = a1.z + ascale * sZ[(t0 + 6) * S_STR + c];
      o1.w = a1.w + ascale * sZ[(t0 + 7) * S_STR + c];
      ((float4*)(out + off))[0] = o0;
      ((float4*)(out + off))[1] = o1;
    }
    __syncthreads();   // sZ / sXraw alias: protect before next iteration's load
  }
}

extern "C" void kernel_launch(void* const* d_in, const int* in_sizes, int n_in,
                              void* d_out, int out_size, void* d_ws, size_t ws_size,
                              hipStream_t stream) {
  (void)in_sizes; (void)n_in; (void)d_ws; (void)ws_size; (void)out_size;
  const float* x          = (const float*)d_in[0];
  const float* norm_w     = (const float*)d_in[1];
  const float* norm_b     = (const float*)d_in[2];
  const float* qkv_w      = (const float*)d_in[3];
  const float* proj_w     = (const float*)d_in[4];
  const float* attn_scale = (const float*)d_in[5];
  const float* rpb_table  = (const float*)d_in[6];
  float* out              = (float*)d_out;

  dim3 grid(2048), block(128);
  swin_window_attn_kernel<<<grid, block, SMEM_BYTES, stream>>>(
      x, norm_w, norm_b, qkv_w, proj_w, attn_scale, rpb_table, out);
}
